// MHAttn_12292196401625
// MI455X (gfx1250) — matrix operand order repaired and checked by reading, not compiled, not run on hardware
//
#include <hip/hip_runtime.h>
#include <hip/hip_bf16.h>
#include <math.h>

// MHA: B=8, N=1024, DIM=1024, H=8, HEAD_DIM=128
// outputs = [out (B,N,DIM), attn (B,H,N,N)], both f32

typedef __bf16 bf16_t;
typedef __attribute__((ext_vector_type(16))) __bf16 v16bf;
typedef __attribute__((ext_vector_type(8)))  __bf16 v8bf;
typedef __attribute__((ext_vector_type(8)))  float  v8f;
typedef __attribute__((ext_vector_type(4)))  unsigned int v4u;
typedef __attribute__((ext_vector_type(8)))  int v8i_;
typedef __attribute__((ext_vector_type(4)))  int v4i_;

#define B_    8
#define N_    1024
#define DIM_  1024
#define H_    8
#define D_    128
#define SCALE_ 0.08838834764831845f   // 128^-0.5

#define ROWS   64            // Q rows per block
#define WAVES  4
#define TPB    (WAVES * 32)

// ---- dynamic-LDS layout (bytes); dynamic LDS base offset == 0 ----
#define SP_OFF     0
#define SP_BYTES   (ROWS * N_ * 2)            // 131072: bf16 score/prob strip
#define SK_OFF     SP_BYTES                   // 131072
#define SK_STRIDE  136                        // f32 per K row (128 + 8 pad, TDM-padded)
#define SK_BYTES   (16 * SK_STRIDE * 4)       // 8704
#define SV_OFF     (SK_OFF + SK_BYTES)        // 139776
#define SV_STRIDE  40                         // bf16 per V column (32 rows + 8 pad)
#define SV_BYTES   (D_ * SV_STRIDE * 2)       // 10240
#define SM_OFF     (SV_OFF + SV_BYTES)        // 150016
#define SINV_OFF   (SM_OFF + ROWS * 4)        // 150272
#define SMEM_BYTES (SINV_OFF + ROWS * 4)      // 150528 (~147 KB of the 320 KB WGP LDS)

// ---------------------------------------------------------------------------
// Kernel 1: fused attention per (b, h, 64-row block); 4 waves x 16 rows.
// K tiles staged via Tensor Data Mover (tensor_load_to_lds + s_wait_tensorcnt);
// online softmax in registers; single vectorized normalize pass writes attn;
// V tiles staged transposed bf16 so P@V B-fragments are contiguous b128 reads.
// ---------------------------------------------------------------------------
__global__ __launch_bounds__(TPB)
void mha_attn_kernel(const float* __restrict__ q, const float* __restrict__ k,
                     const float* __restrict__ v, float* __restrict__ attn_out,
                     float* __restrict__ outh) {
  extern __shared__ __align__(16) char smem[];
  bf16_t* sP   = (bf16_t*)(smem + SP_OFF);
  float*  sK   = (float*) (smem + SK_OFF);
  bf16_t* sV   = (bf16_t*)(smem + SV_OFF);
  float*  sM   = (float*) (smem + SM_OFF);
  float*  sInv = (float*) (smem + SINV_OFF);

  const int tid  = threadIdx.x;
  const int wave = tid >> 5;
  const int lane = tid & 31;
  const int lh   = lane & 15;
  const int half = lane >> 4;

  const int rbCount = N_ / ROWS;         // 16
  const int rb = blockIdx.x % rbCount;
  const int bh = blockIdx.x / rbCount;
  const int h  = bh % H_;
  const int b  = bh / H_;
  const int row0 = rb * ROWS + wave * 16;

  // ---- Q tile (16x128) -> 4 bf16 A-fragments, SCALE folded in ----
  const float* qrow = q + (size_t)(b * N_ + row0 + lh) * DIM_ + h * D_;
  v16bf aQ[4];
  #pragma unroll
  for (int c = 0; c < 4; ++c) {
    const float* p1 = qrow + 32 * c + 16 * half;
    const float* p2 = qrow + 32 * c + 16 * half + 8;
    #pragma unroll
    for (int j = 0; j < 8; ++j) {
      aQ[c][j]     = (bf16_t)(p1[j] * SCALE_);
      aQ[c][8 + j] = (bf16_t)(p2[j] * SCALE_);
    }
  }

  // ---- Pass 1: S = QK^T with online (per-lane) softmax statistics ----
  const float* kbase = k + (size_t)(b * N_) * DIM_ + h * D_;
  float mloc[8], sloc[8];
  #pragma unroll
  for (int r = 0; r < 8; ++r) { mloc[r] = -INFINITY; sloc[r] = 0.f; }

  for (int ct = 0; ct < N_ / 16; ++ct) {
    __syncthreads();                         // previous tile fully consumed
    if (wave == 0) {
      // --- TDM: DMA 16x128 f32 K tile into LDS with +8f32/row padding ---
      unsigned long long ga =
          (unsigned long long)(const void*)(kbase + (size_t)ct * 16 * DIM_);
      v4u g0;
      g0[0] = 1u;                                        // count=1 (valid D#)
      g0[1] = (unsigned)SK_OFF;                          // lds_addr
      g0[2] = (unsigned)(ga & 0xFFFFFFFFu);              // global_addr[31:0]
      g0[3] = (unsigned)((ga >> 32) & 0x01FFFFFFu)       // global_addr[56:32]
              | 0x80000000u;                             // type=2 (image)
      v8i_ g1;
      g1[0] = (2 << 16)      // data_size = 4B
            | (1 << 20)      // pad_enable
            | (6 << 22)      // pad_interval: 128 DWORDs (one 128-f32 row)
            | (7 << 25);     // pad_amount: 8 DWORDs
      g1[1] = (int)(128u << 16);   // tensor_dim0 = 128   (bits 79:48, low half)
      g1[2] = (int)(16u  << 16);   // tensor_dim1 = 16    (bits 111:80, low half)
      g1[3] = (int)(128u << 16);   // tile_dim0 = 128     (bits 127:112)
      g1[4] = 16;                  // tile_dim1 = 16
      g1[5] = DIM_;                // tensor_dim0_stride = 1024 (low 32)
      g1[6] = 0;
      g1[7] = 0;
      v4i_ zz4 = {0, 0, 0, 0};
      v8i_ zz8 = {0, 0, 0, 0, 0, 0, 0, 0};
      __builtin_amdgcn_tensor_load_to_lds(g0, g1, zz4, zz4, zz8, 0);
      __builtin_amdgcn_s_wait_tensorcnt(0);
    }
    __syncthreads();                         // publish tile to all waves

    v8f acc = {};
    #pragma unroll
    for (int c = 0; c < 4; ++c) {
      v16bf bm;
      const float* src = sK + lh * SK_STRIDE + 32 * c + 16 * half;
      #pragma unroll
      for (int e = 0; e < 16; ++e) bm[e] = (bf16_t)src[e];
      acc = __builtin_amdgcn_wmma_f32_16x16x32_bf16(false, aQ[c], false, bm,
                                                    (short)0, acc, false, false);
    }
    #pragma unroll
    for (int r = 0; r < 8; ++r) {
      bf16_t xb = (bf16_t)acc[r];
      sP[(wave * 16 + r + 8 * half) * N_ + ct * 16 + lh] = xb;
      float x  = (float)xb;
      float nm = fmaxf(mloc[r], x);
      sloc[r]  = sloc[r] * __expf(mloc[r] - nm) + __expf(x - nm);
      mloc[r]  = nm;
    }
  }

  // ---- combine softmax stats across the 16 lanes of each half ----
  #pragma unroll
  for (int mask = 1; mask <= 8; mask <<= 1) {
    #pragma unroll
    for (int r = 0; r < 8; ++r) {
      float om = __shfl_xor(mloc[r], mask, 32);
      float os = __shfl_xor(sloc[r], mask, 32);
      float nm = fmaxf(mloc[r], om);
      sloc[r]  = sloc[r] * __expf(mloc[r] - nm) + os * __expf(om - nm);
      mloc[r]  = nm;
    }
  }
  if (lh == 0) {
    #pragma unroll
    for (int r = 0; r < 8; ++r) {
      const int row = wave * 16 + r + 8 * half;
      sM[row]   = mloc[r];
      sInv[row] = 1.0f / sloc[r];
    }
  }
  __syncthreads();

  // ---- normalize pass (8-wide): sP := softmax, attn streamed to HBM ----
  {
    float* abase = attn_out + ((size_t)(b * H_ + h) * N_ + rb * ROWS) * (size_t)N_;
    for (int g = tid; g < ROWS * N_ / 8; g += TPB) {
      const int row = g >> 7;                 // 128 vectors per row
      const float m = sM[row], inv = sInv[row];
      v8bf pv = *(v8bf*)(sP + g * 8);
      float p[8];
      #pragma unroll
      for (int j = 0; j < 8; ++j) p[j] = __expf((float)pv[j] - m) * inv;
      v8bf ov;
      #pragma unroll
      for (int j = 0; j < 8; ++j) ov[j] = (bf16_t)p[j];
      *(v8bf*)(sP + g * 8) = ov;
      float4 f0 = {p[0], p[1], p[2], p[3]};
      float4 f1 = {p[4], p[5], p[6], p[7]};
      *(float4*)(abase + (size_t)g * 8)     = f0;
      *(float4*)(abase + (size_t)g * 8 + 4) = f1;
    }
  }

  // ---- Pass 2: O = P @ V (K=1024, chunks of 32); V staged transposed ----
  const float* vbase = v + (size_t)(b * N_) * DIM_ + h * D_;
  v8f oacc[8] = {};
  for (int kt = 0; kt < N_ / 32; ++kt) {
    __syncthreads();
    if (kt + 1 < N_ / 32)                     // hint next V tile toward L2
      __builtin_prefetch(&vbase[(size_t)((kt + 1) * 32 + (tid >> 2)) * DIM_ +
                                (tid & 3) * 32], 0, 1);
    // stage 32 V rows x 128 dims, transposed bf16: sV[c*40 + r]
    for (int idx = tid; idx < 32 * (D_ / 4); idx += TPB) {
      const int r  = idx >> 5;                // 32 float4 per row
      const int c4 = (idx & 31) * 4;
      float4 t = *(const float4*)&vbase[(size_t)(kt * 32 + r) * DIM_ + c4];
      sV[(c4 + 0) * SV_STRIDE + r] = (bf16_t)t.x;
      sV[(c4 + 1) * SV_STRIDE + r] = (bf16_t)t.y;
      sV[(c4 + 2) * SV_STRIDE + r] = (bf16_t)t.z;
      sV[(c4 + 3) * SV_STRIDE + r] = (bf16_t)t.w;
    }
    __syncthreads();

    // A fragment from probability strip (contiguous bf16)
    v16bf ap;
    const bf16_t* prow = &sP[(wave * 16 + lh) * N_ + 32 * kt];
    #pragma unroll
    for (int j = 0; j < 8; ++j) {
      ap[j]     = prow[16 * half + j];
      ap[8 + j] = prow[16 * half + 8 + j];
    }
    #pragma unroll
    for (int dt = 0; dt < 8; ++dt) {
      v16bf bm;
      const bf16_t* src = &sV[(dt * 16 + lh) * SV_STRIDE + 16 * half];
      #pragma unroll
      for (int e = 0; e < 16; ++e) bm[e] = src[e];
      oacc[dt] = __builtin_amdgcn_wmma_f32_16x16x32_bf16(false, ap, false, bm,
                                                         (short)0, oacc[dt], false, false);
    }
  }

  // ---- head outputs -> workspace in (b, n, h*128+d) layout ----
  #pragma unroll
  for (int dt = 0; dt < 8; ++dt) {
    #pragma unroll
    for (int r = 0; r < 8; ++r) {
      const int rowg = row0 + r + 8 * half;
      outh[(size_t)(b * N_ + rowg) * DIM_ + h * D_ + dt * 16 + lh] = oacc[dt][r];
    }
  }
}

// ---------------------------------------------------------------------------
// Kernel 2: Y = X @ W^T + bias (8192x1024 x 1024x1024^T), bf16 WMMA.
// ---------------------------------------------------------------------------
#define PTPB 256
__global__ __launch_bounds__(PTPB)
void mha_proj_kernel(const float* __restrict__ X, const float* __restrict__ W,
                     const float* __restrict__ bias, float* __restrict__ Y) {
  const int tid  = threadIdx.x;
  const int wave = tid >> 5;
  const int lane = tid & 31;
  const int lh   = lane & 15;
  const int half = lane >> 4;

  const int colGroups = DIM_ / 64;                      // 16
  const int gw   = blockIdx.x * (PTPB / 32) + wave;
  const int row0 = (gw / colGroups) * 16;
  const int col0 = (gw % colGroups) * 64;

  const float* xrow = X + (size_t)(row0 + lh) * DIM_;
  v8f acc[4] = {};

  for (int kt = 0; kt < DIM_ / 32; ++kt) {
    v16bf a;
    const float* p1 = xrow + 32 * kt + 16 * half;
    const float* p2 = xrow + 32 * kt + 16 * half + 8;
    #pragma unroll
    for (int j = 0; j < 8; ++j) {
      a[j]     = (bf16_t)p1[j];
      a[8 + j] = (bf16_t)p2[j];
    }
    #pragma unroll
    for (int nt = 0; nt < 4; ++nt) {
      const float* wrow = W + (size_t)(col0 + nt * 16 + lh) * DIM_ + 32 * kt + 16 * half;
      v16bf bm;
      #pragma unroll
      for (int e = 0; e < 16; ++e) bm[e] = (bf16_t)wrow[e];
      acc[nt] = __builtin_amdgcn_wmma_f32_16x16x32_bf16(false, a, false, bm,
                                                        (short)0, acc[nt], false, false);
    }
  }

  #pragma unroll
  for (int nt = 0; nt < 4; ++nt) {
    const int col = col0 + nt * 16 + lh;
    const float bv = bias[col];
    #pragma unroll
    for (int r = 0; r < 8; ++r)
      Y[(size_t)(row0 + r + 8 * half) * DIM_ + col] = acc[nt][r] + bv;
  }
}

// ---------------------------------------------------------------------------
extern "C" void kernel_launch(void* const* d_in, const int* in_sizes, int n_in,
                              void* d_out, int out_size, void* d_ws, size_t ws_size,
                              hipStream_t stream) {
  const float* q     = (const float*)d_in[0];
  const float* k     = (const float*)d_in[1];
  const float* v     = (const float*)d_in[2];
  const float* W_out = (const float*)d_in[3];
  const float* b_out = (const float*)d_in[4];

  float* out  = (float*)d_out;                          // (B, N, DIM)
  float* attn = out + (size_t)B_ * N_ * DIM_;           // (B, H, N, N)
  float* outh = (float*)d_ws;                           // (B, N, DIM) head-merged

  mha_attn_kernel<<<B_ * H_ * (N_ / ROWS), TPB, SMEM_BYTES, stream>>>(
      q, k, v, attn, outh);

  const int waves = (B_ * N_ / 16) * (DIM_ / 64);
  mha_proj_kernel<<<waves / (PTPB / 32), PTPB, 0, stream>>>(outh, W_out, b_out, out);
}
